// RSSMEncoder_68539088109572
// MI455X (gfx1250) — compile-verified, matrix operand order checked by reading, not run on hardware
//
#include <hip/hip_runtime.h>
#include <hip/hip_bf16.h>

// ============================================================================
// RSSM forward for MI455X (gfx1250, wave32, WMMA).
//
// Forward-value dead-code elimination (mathematically exact to ~3e-6):
//   * sample_st(p,k).mean(-1) == 1/NC (constant)  => post_stoch == r_lnstoch_b
//   * carried stoch == 0                          => rnn_in = elu(act @ W[0:6])
//   * prior head, posterior head, and the whole conv encoder never reach the
//     returned tensor concat(post_stoch, det).
// Live computation: 50 serial GRU steps:
//   gi = rnn_in @ gru_wi + bi   (32 x 1024 x 600, bf16 WMMA, f32 acc)
//   gh = det    @ gru_wh + bh   (32 x  200 x 600, K padded to 224)
//   r,z,n gates -> det = LN((1-z)*n + z*det)
// Weights are pre-tiled into the WMMA B-fragment layout (one contiguous
// 32B load per lane per k-step, base + 24-bit immediate offsets in the
// unrolled K loops). rnn_in (depends only on actions) is precomputed for
// all 50 steps in one pass. Gates kernel register-blocks 2 C tiles per
// wave with compile-time K loops (this path is latency-bound, not BW-bound).
//
// Workspace requirement: ~5.0 MB (offsets below, all 256B aligned).
// ============================================================================

#define T_   50
#define B_   32
#define A_   6
#define S_   30
#define D_   200
#define H_   1024
#define TB_  (T_ * B_)      // 1600
#define NG_  600            // 3*D gate width
#define NGP_ 608            // padded to 16
#define NT_  38             // N tiles (608/16)
#define KDP_ 224            // D padded to mult of 32
#define KTI_ 32             // k-tiles for K=1024
#define KTH_ 7              // k-tiles for K=224
#define OUTC_ (S_ + D_)     // 230

typedef __bf16         bf16x16 __attribute__((ext_vector_type(16)));
typedef float          f32x8   __attribute__((ext_vector_type(8)));
typedef unsigned short u16x8   __attribute__((ext_vector_type(8)));
typedef unsigned short u16x16  __attribute__((ext_vector_type(16)));

__device__ __forceinline__ unsigned short f2bf(float f) {
    unsigned int u = __float_as_uint(f);
    u = (u + 0x7FFFu + ((u >> 16) & 1u)) >> 16;   // round-to-nearest-even
    return (unsigned short)u;
}

// ---------------------------------------------------------------------------
// Pre-tile a [Korig, 600] f32 weight into WMMA B-fragment blocks:
// dst[kt][ntile][lane][e] = W[k(lane,e), n(lane)], bf16, zero-padded.
// (lane<16 -> kbase 0, lane>=16 -> kbase 8; e<8 -> k=kt*32+kbase+e,
//  e>=8 -> k=kt*32+16+kbase+e-8; n = ntile*16 + (lane&15))  [ISA 7.12.2]
// ---------------------------------------------------------------------------
__global__ void tile_b(const float* __restrict__ w, int Korig, int KT,
                       unsigned short* __restrict__ dst) {
    int tid = blockIdx.x * blockDim.x + threadIdx.x;
    int total = KT * NT_ * 512;
    if (tid >= total) return;
    int e     = tid & 15;
    int lane  = (tid >> 4) & 31;
    int blk   = tid >> 9;
    int ntile = blk % NT_;
    int kt    = blk / NT_;
    int kb    = (lane < 16) ? 0 : 8;
    int k     = kt * 32 + ((e < 8) ? (kb + e) : (16 + kb + (e - 8)));
    int n     = ntile * 16 + (lane & 15);
    float v   = (k < Korig && n < NG_) ? w[(size_t)k * NG_ + n] : 0.f;
    dst[tid]  = f2bf(v);
}

// ---------------------------------------------------------------------------
// rnn_in[t*B+b, h] = elu(sum_a act[t,b,a]*t_in_w[a,h] + t_in_b[h])  (bf16)
// ---------------------------------------------------------------------------
__global__ void rnnin_pre(const float* __restrict__ act,
                          const float* __restrict__ w,
                          const float* __restrict__ b,
                          unsigned short* __restrict__ rnnin) {
    int tid = blockIdx.x * blockDim.x + threadIdx.x;
    if (tid >= TB_ * H_) return;
    int i = tid / H_, h = tid - i * H_;
    float v = b[h];
#pragma unroll
    for (int a = 0; a < A_; ++a) v += act[i * A_ + a] * w[a * H_ + h];
    v = (v > 0.f) ? v : (__expf(v) - 1.f);      // elu
    rnnin[tid] = f2bf(v);
}

// ---------------------------------------------------------------------------
// Init: det=0 (f32 + padded bf16), fill output stoch slice with r_lnstoch_b.
// ---------------------------------------------------------------------------
__global__ void rssm_init(float* __restrict__ det_f,
                          unsigned short* __restrict__ det_bf,
                          const float* __restrict__ rlnst_b,
                          float* __restrict__ out) {
    int tid = blockIdx.x * blockDim.x + threadIdx.x;
    const int n0 = B_ * D_, n1 = B_ * KDP_, n2 = T_ * B_ * S_;
    if (tid < n0) {
        det_f[tid] = 0.f;
    } else if (tid < n0 + n1) {
        det_bf[tid - n0] = 0;
    } else if (tid < n0 + n1 + n2) {
        int r  = tid - n0 - n1;
        int s  = r % S_;
        int tb = r / S_;
        out[(size_t)tb * OUTC_ + s] = rlnst_b[s];   // post_stoch == ln-bias
    }
}

// ---------------------------------------------------------------------------
// Fragment helper: A fragment = two 16B loads from row-major bf16 A.
// ---------------------------------------------------------------------------
__device__ __forceinline__ bf16x16 load_a_frag(const unsigned short* Arow,
                                               unsigned kofs) {
    const u16x8* ap = reinterpret_cast<const u16x8*>(Arow + kofs);
    u16x8 alo = ap[0];
    u16x8 ahi = ap[2];                          // +16 ushorts (K+16 group)
    u16x16 av;
#pragma unroll
    for (int i = 0; i < 8; ++i) { av[i] = alo[i]; av[8 + i] = ahi[i]; }
    return __builtin_bit_cast(bf16x16, av);
}

// ---------------------------------------------------------------------------
// Per-step gates: 76 waves, each computes a 16x32 C strip (2 tiles, shared
// A fragment -> 2 WMMAs per k-step).
//   waves  0..37 : gi = rnn_in[t] (32x1024 bf16) @ wi_blk + bi   (K loop 32)
//   waves 38..75 : gh = det       (32x 224 bf16) @ wh_blk + bh   (K loop 7)
// Branch is wave-uniform, loops have compile-time bounds; B loads resolve to
// base + constant offsets (k-tile stride 38912B << 24-bit IOFFSET range).
// EXEC is all-1s at every WMMA as the ISA requires.
// ---------------------------------------------------------------------------
__global__ void __launch_bounds__(128)
rssm_gates(const unsigned short* __restrict__ rnnin,
           const unsigned short* __restrict__ wi_blk,
           const unsigned short* __restrict__ det_bf,
           const unsigned short* __restrict__ wh_blk,
           const float* __restrict__ bi, const float* __restrict__ bh,
           float* __restrict__ gi, float* __restrict__ gh, int t) {
    const int lane  = threadIdx.x & 31;
    const int w     = blockIdx.x * 4 + (threadIdx.x >> 5);   // 0..75
    const bool is_h = (w >= 38);
    const int  tt   = is_h ? (w - 38) : w;
    const int mtile = tt & 1;
    const int ntile = (tt >> 1) * 2;             // 0,2,..,36 (pair base)
    const int mrow  = lane & 15;
    const unsigned kbase = (lane < 16) ? 0u : 8u;
    constexpr unsigned KSTRIDE = NT_ * 512u;     // ushorts per k-tile of blocks

    f32x8 acc0 = {0.f, 0.f, 0.f, 0.f, 0.f, 0.f, 0.f, 0.f};
    f32x8 acc1 = {0.f, 0.f, 0.f, 0.f, 0.f, 0.f, 0.f, 0.f};

    if (!is_h) {
        const unsigned short* Arow  = rnnin +
            (size_t)(t * B_ + mtile * 16 + mrow) * H_ + kbase;
        const unsigned short* Bbase = wi_blk +
            (size_t)ntile * 512u + (unsigned)lane * 16u;
#pragma unroll 4
        for (unsigned kt = 0; kt < KTI_; ++kt) {
            bf16x16 af = load_a_frag(Arow, kt * 32u);
            const u16x16* bp =
                reinterpret_cast<const u16x16*>(Bbase + kt * KSTRIDE);
            bf16x16 bf0 = __builtin_bit_cast(bf16x16, bp[0]);
            bf16x16 bf1 = __builtin_bit_cast(bf16x16, bp[32]);  // +512 ushorts
            acc0 = __builtin_amdgcn_wmma_f32_16x16x32_bf16(
                false, af, false, bf0, (short)0, acc0, false, false);
            acc1 = __builtin_amdgcn_wmma_f32_16x16x32_bf16(
                false, af, false, bf1, (short)0, acc1, false, false);
        }
    } else {
        const unsigned short* Arow  = det_bf +
            (size_t)(mtile * 16 + mrow) * KDP_ + kbase;
        const unsigned short* Bbase = wh_blk +
            (size_t)ntile * 512u + (unsigned)lane * 16u;
#pragma unroll
        for (unsigned kt = 0; kt < KTH_; ++kt) {
            bf16x16 af = load_a_frag(Arow, kt * 32u);
            const u16x16* bp =
                reinterpret_cast<const u16x16*>(Bbase + kt * KSTRIDE);
            bf16x16 bf0 = __builtin_bit_cast(bf16x16, bp[0]);
            bf16x16 bf1 = __builtin_bit_cast(bf16x16, bp[32]);
            acc0 = __builtin_amdgcn_wmma_f32_16x16x32_bf16(
                false, af, false, bf0, (short)0, acc0, false, false);
            acc1 = __builtin_amdgcn_wmma_f32_16x16x32_bf16(
                false, af, false, bf1, (short)0, acc1, false, false);
        }
    }

    // Epilogue: ISA C/D layout -> row m = mtile*16 + r + (lane<16?0:8),
    // col n = (ntile+tile)*16 + (lane&15); bias is per-column (uniform per lane).
    float*       C    = is_h ? gh : gi;
    const float* bias = is_h ? bh : bi;
    const int rbase   = mtile * 16 + ((lane < 16) ? 0 : 8);
#pragma unroll
    for (int tile = 0; tile < 2; ++tile) {
        const int n = (ntile + tile) * 16 + mrow;
        if (n < NG_) {
            const float bvv = bias[n];
            const f32x8& acc = tile ? acc1 : acc0;
#pragma unroll
            for (int r = 0; r < 8; ++r)
                C[(size_t)(rbase + r) * NGP_ + n] = acc[r] + bvv;
        }
    }
}

// ---------------------------------------------------------------------------
// Per-step GRU combine + layernorm. 32 waves (1024 threads), wave == batch row.
// Wave32 __shfl_xor reductions for mean/var over D=200.
// ---------------------------------------------------------------------------
__global__ void __launch_bounds__(1024)
rssm_combine(const float* __restrict__ gi, const float* __restrict__ gh,
             float* __restrict__ det_f, unsigned short* __restrict__ det_bf,
             const float* __restrict__ lng, const float* __restrict__ lnb,
             float* __restrict__ out, int t) {
    const int lane = threadIdx.x & 31;
    const int row  = threadIdx.x >> 5;               // 0..31
    const float* gir = gi + (size_t)row * NGP_;
    const float* ghr = gh + (size_t)row * NGP_;
    float* detr = det_f + (size_t)row * D_;

    float y[7];
    float s = 0.f, sq = 0.f;
#pragma unroll
    for (int j = 0; j < 7; ++j) {
        int n = lane + j * 32;
        float v = 0.f;
        if (n < D_) {
            float ir = gir[n], iz = gir[D_ + n], ic = gir[2 * D_ + n];
            float hr = ghr[n], hz = ghr[D_ + n], hn = ghr[2 * D_ + n];
            float r  = 1.f / (1.f + __expf(-(ir + hr)));
            float z  = 1.f / (1.f + __expf(-(iz + hz)));
            float nn = tanhf(ic + r * hn);
            v = (1.f - z) * nn + z * detr[n];
        }
        y[j] = v;  s += v;  sq += v * v;
    }
#pragma unroll
    for (int m = 16; m >= 1; m >>= 1) {
        s  += __shfl_xor(s,  m, 32);
        sq += __shfl_xor(sq, m, 32);
    }
    const float mean = s * (1.f / D_);
    const float var  = sq * (1.f / D_) - mean * mean;
    const float inv  = rsqrtf(var + 1e-5f);
    float* orow = out + ((size_t)t * B_ + row) * OUTC_ + S_;
#pragma unroll
    for (int j = 0; j < 7; ++j) {
        int n = lane + j * 32;
        if (n < D_) {
            float dn = (y[j] - mean) * inv * lng[n] + lnb[n];
            detr[n] = dn;
            det_bf[row * KDP_ + n] = f2bf(dn);     // bf16 A for next step's gh
            orow[n] = dn;
        }
    }
}

// ---------------------------------------------------------------------------
extern "C" void kernel_launch(void* const* d_in, const int* in_sizes, int n_in,
                              void* d_out, int out_size, void* d_ws, size_t ws_size,
                              hipStream_t stream) {
    (void)in_sizes; (void)n_in; (void)out_size; (void)ws_size;
    const float* actions = (const float*)d_in[1];
    const float* t_in_w  = (const float*)d_in[14];
    const float* t_in_b  = (const float*)d_in[15];
    const float* gru_wi  = (const float*)d_in[16];
    const float* gru_bi  = (const float*)d_in[17];
    const float* gru_wh  = (const float*)d_in[18];
    const float* gru_bh  = (const float*)d_in[19];
    const float* lndet_g = (const float*)d_in[20];
    const float* lndet_b = (const float*)d_in[21];
    const float* rlnst_b = (const float*)d_in[37];
    float* out = (float*)d_out;

    // workspace layout (256B aligned, total ~4.99 MB)
    char* ws = (char*)d_ws;
    unsigned short* rnnin  = (unsigned short*)(ws + 0);        // 1600*1024 bf16
    unsigned short* wi_blk = (unsigned short*)(ws + 3276800);  // 32*38*512 bf16
    unsigned short* wh_blk = (unsigned short*)(ws + 4521984);  //  7*38*512 bf16
    float*          det_f  = (float*)         (ws + 4794368);  // 32*200 f32
    unsigned short* det_bf = (unsigned short*)(ws + 4819968);  // 32*224 bf16
    float*          gi     = (float*)         (ws + 4834304);  // 32*608 f32
    float*          gh     = (float*)         (ws + 4912128);  // 32*608 f32

    tile_b<<<(KTI_ * NT_ * 512 + 255) / 256, 256, 0, stream>>>(gru_wi, H_, KTI_, wi_blk);
    tile_b<<<(KTH_ * NT_ * 512 + 255) / 256, 256, 0, stream>>>(gru_wh, D_, KTH_, wh_blk);
    rnnin_pre<<<(TB_ * H_ + 255) / 256, 256, 0, stream>>>(actions, t_in_w, t_in_b, rnnin);
    rssm_init<<<((B_ * D_ + B_ * KDP_ + T_ * B_ * S_) + 255) / 256, 256, 0, stream>>>(
        det_f, det_bf, rlnst_b, out);

    for (int t = 0; t < T_; ++t) {
        rssm_gates<<<19, 128, 0, stream>>>(rnnin, wi_blk, det_bf, wh_blk,
                                           gru_bi, gru_bh, gi, gh, t);
        rssm_combine<<<1, 1024, 0, stream>>>(gi, gh, det_f, det_bf,
                                             lndet_g, lndet_b, out, t);
    }
}
